// SingleLayerGRU_87823491268729
// MI455X (gfx1250) — compile-verified, matrix operand order checked by reading, not run on hardware
//
#include <hip/hip_runtime.h>
#include <hip/hip_bf16.h>
#include <stdint.h>

// ---------------------------------------------------------------------------
// GRU recurrence on MI455X (gfx1250):
//   4 batch tiles (M=16) x 16 column groups = 64 persistent workgroups.
//   Each WG keeps its f16 weight slice resident in LDS (WGP has 320KB).
//   Per step: WMMA fz -> sigmoid -> g=f*x exchange -> WMMA r -> tanh ->
//   x update exchange. Tile-scoped global atomic barriers (16 WGs each).
//   TDM (tensor_load_to_lds) refills LDS copies of g and x each step.
// ---------------------------------------------------------------------------

typedef _Float16 f16;
typedef __attribute__((ext_vector_type(16))) _Float16 v16h;
typedef __attribute__((ext_vector_type(8)))  _Float16 v8h;
typedef __attribute__((ext_vector_type(8)))  float    v8f;
typedef __attribute__((ext_vector_type(4)))  unsigned int v4u;
typedef __attribute__((ext_vector_type(8)))  int v8i;
typedef __attribute__((ext_vector_type(4)))  int v4i;

constexpr int kB   = 64;
constexpr int kT   = 2048;
constexpr int kIN  = 64;
constexpr int kH   = 512;
constexpr int kOUT = 32;
constexpr int kK   = kIN + kH;      // 576

// LDS layout (units: f16 halves). All region bases 16B aligned.
constexpr int O_WFZ = 0;                    // 64 cols x 576 K   (fz slice, N-major)
constexpr int O_WR  = O_WFZ + 64 * kK;      // 32 cols x 576 K   (r slice,  N-major)
constexpr int O_X   = O_WR  + 32 * kK;      // 16 x 512 current state x (full H)
constexpr int O_G   = O_X   + 16 * kH;      // 16 x 512 g = f*x (full H)
constexpr int O_U   = O_G   + 16 * kH;      // 16 x 64 u_t
constexpr int O_END = O_U   + 16 * kIN;     // 72704 halves = 145408 bytes

// Workspace layout (bytes)
constexpr size_t WS_WFZ  = 0;                                   // 1024*576 f16
constexpr size_t WS_WR   = WS_WFZ  + (size_t)1024 * kK * 2;     // 512*576 f16
constexpr size_t WS_WOUT = WS_WR   + (size_t)512  * kK * 2;     // 32*512 f16
constexpr size_t WS_XSH  = WS_WOUT + (size_t)kOUT * kH * 2;     // 64*512 f16
constexpr size_t WS_GSH  = WS_XSH  + (size_t)kB   * kH * 2;     // 64*512 f16
constexpr size_t WS_CNT  = WS_GSH  + (size_t)kB   * kH * 2;     // 4 counters

__device__ inline float sigf(float x)  { return 1.0f / (1.0f + __expf(-x)); }
__device__ inline float tanhf_fast(float x) {
  float e = __expf(2.0f * x);
  return (e - 1.0f) / (e + 1.0f);
}

// Load a 16x32 f16 fragment per ISA 7.12.2 striping: lane holds row (lane&15),
// two 16-byte K chunks at K = ks*8 and K = ks*8+16 (ks = lane>>4).
// B operands use the same gather over N-major (transposed) weight storage.
__device__ inline v16h ldfrag(const f16* base, int stride_h, int lane) {
  const int m  = lane & 15;
  const int ks = (lane >> 4) & 1;
  const f16* p = base + m * stride_h + ks * 8;
  v8h c0 = *(const v8h*)(p);
  v8h c1 = *(const v8h*)(p + 16);
  v16h r;
#pragma unroll
  for (int i = 0; i < 8; ++i) { r[i] = c0[i]; r[i + 8] = c1[i]; }
  return r;
}

// 1-D Tensor Data Mover copy: global -> LDS, nelem 2-byte elements.
// D# per CDNA5 ISA (group0: count/lds/global/type,
// group1: data_size=2B, dim0 = stride0 = nelem, tile_dim0 = nelem).
// This toolchain's builtin takes 6 args (g0 v4u, g1 v8i, g2 v4i, g3 v4i,
// extra v8i group, cpol).
__device__ inline void tdm_load_1d(unsigned lds_byte_off, const void* gptr,
                                   unsigned nelem) {
  unsigned long long ga = (unsigned long long)gptr;
  v4u g0;
  g0[0] = 1u;                                        // count=1, user descriptor
  g0[1] = lds_byte_off;                              // lds_addr
  g0[2] = (unsigned)(ga & 0xffffffffu);              // global_addr[31:0]
  g0[3] = (unsigned)((ga >> 32) & 0x1ffffffu) | (2u << 30); // addr[56:32], type=2
  v8i g1;
  g1[0] = (int)(1u << 16);                           // data_size=1 (2 bytes)
  g1[1] = (int)((nelem & 0xffffu) << 16);            // tensor_dim0[15:0]
  g1[2] = (int)((nelem >> 16) & 0xffffu) | (1 << 16);// tensor_dim0[31:16], dim1=1
  g1[3] = (int)((nelem & 0xffffu) << 16);            // tile_dim0 (<=65535 here)
  g1[4] = 1;                                         // tile_dim1=1
  g1[5] = (int)nelem;                                // tensor_dim0_stride[31:0]
  g1[6] = 0;
  g1[7] = 0;
  v4i z4 = {0, 0, 0, 0};
  v8i z8 = {0, 0, 0, 0, 0, 0, 0, 0};
  __builtin_amdgcn_tensor_load_to_lds(g0, g1, z4, z4, z8, 0);
}

// ---------------------------------------------------------------------------
// Prep: fp32 weights -> f16, transposed to N-major; x0 -> f16; zero barriers.
// ---------------------------------------------------------------------------
__global__ void gru_prep_87823491268729(const float* __restrict__ kfz,
                                        const float* __restrict__ kr,
                                        const float* __restrict__ wout,
                                        const float* __restrict__ x0,
                                        f16* wfzT, f16* wrT, f16* woutT,
                                        f16* xsh, unsigned* cnt) {
  size_t gid    = (size_t)blockIdx.x * blockDim.x + threadIdx.x;
  size_t stride = (size_t)gridDim.x * blockDim.x;
  for (size_t i = gid; i < (size_t)(2 * kH) * kK; i += stride) {
    size_t n = i / kK, k = i % kK;
    wfzT[i] = (f16)kfz[k * (2 * kH) + n];
  }
  for (size_t i = gid; i < (size_t)kH * kK; i += stride) {
    size_t n = i / kK, k = i % kK;
    wrT[i] = (f16)kr[k * kH + n];
  }
  for (size_t i = gid; i < (size_t)kOUT * kH; i += stride) {
    woutT[i] = (f16)wout[i];                 // W_out already (OUT,H) = N-major
  }
  for (size_t i = gid; i < (size_t)kB * kH; i += stride) {
    xsh[i] = (f16)x0[i];                     // x0 is (B,1,H) flat
  }
  if (gid < 8) cnt[gid] = 0;
}

// ---------------------------------------------------------------------------
// Main persistent kernel: grid = 64 WGs x 256 threads (8 wave32).
// blockIdx = tile*16 + cg; tile in [0,4) batch rows, cg in [0,16) H columns.
// ---------------------------------------------------------------------------
__global__ __launch_bounds__(256) void gru_main_87823491268729(
    const float* __restrict__ u, const float* __restrict__ bias_fz,
    const float* __restrict__ bias_r, const float* __restrict__ b_out,
    const f16* __restrict__ wfzT, const f16* __restrict__ wrT,
    const f16* __restrict__ woutT, f16* xsh, f16* gsh, unsigned* cnt,
    float* __restrict__ out) {
  const int tid  = threadIdx.x;
  const int lane = tid & 31;
  const int wave = tid >> 5;
  const int tile = blockIdx.x >> 4;   // batch tile (rows tile*16 .. +16)
  const int cg   = blockIdx.x & 15;   // column group (h cols cg*32 .. +32)

  __shared__ __attribute__((aligned(128))) f16 smem[O_END];
  __shared__ float sZ[16 * 32];       // z gate values for this WG's 32 columns

  // ---- init: TDM weight slices + initial x into LDS (resident all steps) ----
  if (wave == 0) {
    tdm_load_1d((unsigned)(O_WFZ * 2), wfzT + (size_t)cg * 64 * kK, 64 * kK);
    tdm_load_1d((unsigned)(O_WR * 2),  wrT  + (size_t)cg * 32 * kK, 32 * kK);
    tdm_load_1d((unsigned)(O_X * 2),   xsh  + (size_t)tile * 16 * kH, 16 * kH);
    __builtin_amdgcn_s_wait_tensorcnt(0);
  }
  __syncthreads();

  unsigned phase = 0;
  unsigned* bar  = cnt + tile;
  const int nl = lane & 15;
  const int ks = (lane >> 4) & 1;

  for (int t = 0; t < kT; ++t) {
    // ---- stage u_t (fp32 global -> f16 LDS), 16x64 ----
#pragma unroll
    for (int e = tid; e < 16 * kIN; e += 256) {
      int m = e >> 6, i = e & 63;
      float v = u[(((size_t)(tile * 16 + m)) * kT + t) * kIN + i];
      smem[O_U + m * kIN + i] = (f16)v;
    }
    __syncthreads();

    if (wave < 4) {
      // ---- fz = sigmoid([u,x] @ Wfz + b): this WG's 64 columns, 4 N-tiles ----
      v8f acc = {};
      const f16* wb = smem + O_WFZ + wave * 16 * kK;
#pragma unroll
      for (int kt = 0; kt < kK / 32; ++kt) {
        int k0 = kt * 32;
        v16h a = (k0 < kIN) ? ldfrag(smem + O_U + k0, kIN, lane)
                            : ldfrag(smem + O_X + (k0 - kIN), kH, lane);
        v16h b = ldfrag(wb + k0, kK, lane);
        acc = __builtin_amdgcn_wmma_f32_16x16x32_f16(false, a, false, b,
                                                     (short)0, acc, false, false);
      }
      if (wave < 2) {
        // f gate -> g = f*x, publish g columns to L2-resident exchange buffer
        int h = cg * 32 + wave * 16 + nl;
        float bia = bias_fz[h];
        f16* gdst = gsh + (size_t)tile * 16 * kH + h;
#pragma unroll
        for (int i = 0; i < 8; ++i) {
          int M = i + 8 * ks;
          float fv = sigf(acc[i] + bia);
          float gv = fv * (float)smem[O_X + M * kH + h];
          gdst[(size_t)M * kH] = (f16)gv;
        }
      } else {
        // z gate -> LDS for the state update
        int hl = (wave - 2) * 16 + nl;
        float bia = bias_fz[kH + cg * 32 + hl];
#pragma unroll
        for (int i = 0; i < 8; ++i) {
          int M = i + 8 * ks;
          sZ[M * 32 + hl] = sigf(acc[i] + bia);
        }
      }
    } else if (cg == 0 && wave < 6) {
      // ---- y_t = x_t @ Wout^T + b (pre-update x); overlapped with fz ----
      int nt = wave - 4;
      v8f acc = {};
      const f16* wb = woutT + (size_t)(nt * 16) * kH;
#pragma unroll
      for (int kt = 0; kt < kH / 32; ++kt) {
        v16h a = ldfrag(smem + O_X + kt * 32, kH, lane);
        v16h b = ldfrag(wb + kt * 32, kH, lane);
        acc = __builtin_amdgcn_wmma_f32_16x16x32_f16(false, a, false, b,
                                                     (short)0, acc, false, false);
      }
      float bia = b_out[nt * 16 + nl];
#pragma unroll
      for (int i = 0; i < 8; ++i) {
        int M = i + 8 * ks;
        out[(((size_t)(tile * 16 + M)) * kT + t) * kOUT + nt * 16 + nl] =
            acc[i] + bia;
      }
    }

    // ---- tile-group barrier #1 (g published), then TDM-refill full g ----
    __syncthreads();
    if (tid == 0) {
      __threadfence();
      __hip_atomic_fetch_add(bar, 1u, __ATOMIC_RELEASE, __HIP_MEMORY_SCOPE_AGENT);
      unsigned tgt = 16u * (phase + 1);
      while (__hip_atomic_load(bar, __ATOMIC_ACQUIRE, __HIP_MEMORY_SCOPE_AGENT) < tgt)
        __builtin_amdgcn_s_sleep(1);
    }
    phase++;
    __syncthreads();
    if (wave == 0) {
      tdm_load_1d((unsigned)(O_G * 2), gsh + (size_t)tile * 16 * kH, 16 * kH);
      __builtin_amdgcn_s_wait_tensorcnt(0);
    }
    __syncthreads();

    if (wave < 2) {
      // ---- r = tanh([u,g] @ Wr + b); x' = (1-z)x + z r  (our 32 columns) ----
      v8f acc = {};
      const f16* wb = smem + O_WR + wave * 16 * kK;
#pragma unroll
      for (int kt = 0; kt < kK / 32; ++kt) {
        int k0 = kt * 32;
        v16h a = (k0 < kIN) ? ldfrag(smem + O_U + k0, kIN, lane)
                            : ldfrag(smem + O_G + (k0 - kIN), kH, lane);
        v16h b = ldfrag(wb + k0, kK, lane);
        acc = __builtin_amdgcn_wmma_f32_16x16x32_f16(false, a, false, b,
                                                     (short)0, acc, false, false);
      }
      int hl = wave * 16 + nl;
      int h  = cg * 32 + hl;
      float bia = bias_r[h];
      f16* xdst = xsh + (size_t)tile * 16 * kH + h;
#pragma unroll
      for (int i = 0; i < 8; ++i) {
        int M = i + 8 * ks;
        float rv = tanhf_fast(acc[i] + bia);
        float zz = sZ[M * 32 + hl];
        float xo = (float)smem[O_X + M * kH + h];
        xdst[(size_t)M * kH] = (f16)((1.0f - zz) * xo + zz * rv);
      }
    }

    // ---- tile-group barrier #2 (x published), then TDM-refill full x ----
    __syncthreads();
    if (tid == 0) {
      __threadfence();
      __hip_atomic_fetch_add(bar, 1u, __ATOMIC_RELEASE, __HIP_MEMORY_SCOPE_AGENT);
      unsigned tgt = 16u * (phase + 1);
      while (__hip_atomic_load(bar, __ATOMIC_ACQUIRE, __HIP_MEMORY_SCOPE_AGENT) < tgt)
        __builtin_amdgcn_s_sleep(1);
    }
    phase++;
    __syncthreads();
    if (wave == 0) {
      tdm_load_1d((unsigned)(O_X * 2), xsh + (size_t)tile * 16 * kH, 16 * kH);
      __builtin_amdgcn_s_wait_tensorcnt(0);
    }
    __syncthreads();
  }
}

// ---------------------------------------------------------------------------
extern "C" void kernel_launch(void* const* d_in, const int* in_sizes, int n_in,
                              void* d_out, int out_size, void* d_ws, size_t ws_size,
                              hipStream_t stream) {
  (void)in_sizes; (void)n_in; (void)out_size; (void)ws_size;
  const float* u    = (const float*)d_in[0];
  const float* x0   = (const float*)d_in[1];
  const float* kfz  = (const float*)d_in[2];
  const float* bfz  = (const float*)d_in[3];
  const float* kr   = (const float*)d_in[4];
  const float* br   = (const float*)d_in[5];
  const float* wout = (const float*)d_in[6];
  const float* bo   = (const float*)d_in[7];

  char* ws = (char*)d_ws;
  f16* wfzT  = (f16*)(ws + WS_WFZ);
  f16* wrT   = (f16*)(ws + WS_WR);
  f16* woutT = (f16*)(ws + WS_WOUT);
  f16* xsh   = (f16*)(ws + WS_XSH);
  f16* gsh   = (f16*)(ws + WS_GSH);
  unsigned* cnt = (unsigned*)(ws + WS_CNT);

  gru_prep_87823491268729<<<512, 256, 0, stream>>>(kfz, kr, wout, x0, wfzT, wrT,
                                                   woutT, xsh, cnt);
  gru_main_87823491268729<<<64, 256, 0, stream>>>(u, bfz, br, bo, wfzT, wrT,
                                                  woutT, xsh, gsh, cnt,
                                                  (float*)d_out);
}
